// SelfChannelAttentionModule_26499948216894
// MI455X (gfx1250) — matrix operand order repaired and checked by reading, not compiled, not run on hardware
//
#include <hip/hip_runtime.h>
#include <hip/hip_bf16.h>
#include <math.h>

// ---------------------------------------------------------------------------
// SelfChannelAttention on MI455X (gfx1250, wave32)
//   GEMM core: 256 threads = 8 waves (4 M x 2 N), each wave owns a 32x32
//   output tile = 2x2 v_wmma_f32_16x16x32_f16 with fragment reuse.
//   Block tile 128(M) x 64(N), K stepped by 32 through LDS.
//   f16-source tiles staged with CDNA5 async-to-LDS (ASYNCcnt) copies.
// ---------------------------------------------------------------------------

#define C_DIM 256
#define HW    3136   // 56*56
#define PHW   784    // 28*28
#define B_DIM 32
#define BM    128
#define BN    64
#define LDSB_U32 (BM * 17)              // ldsB starts after ldsA
#define LDS_TOT  (BM * 17 + BN * 17)

typedef __attribute__((ext_vector_type(16))) _Float16 v16h;
typedef __attribute__((ext_vector_type(8)))  float    v8f;

union Frag { unsigned u[8]; v16h v; };

__device__ __forceinline__ unsigned pack_f32x2_h2(float a, float b) {
    union { _Float16 h[2]; unsigned u; } p;
    p.h[0] = (_Float16)a; p.h[1] = (_Float16)b;
    return p.u;
}
__device__ __forceinline__ unsigned pack_h2_raw(unsigned short a, unsigned short b) {
    return (unsigned)a | ((unsigned)b << 16);
}

// CDNA5 async global->LDS copy, 4 bytes per lane, tracked by ASYNCcnt.
__device__ __forceinline__ void async_ld_b32(unsigned lds_byte_off, unsigned gbyte_off,
                                             unsigned long long sbase) {
    asm volatile("global_load_async_to_lds_b32 %0, %1, %2"
                 :: "v"(lds_byte_off), "v"(gbyte_off), "s"(sbase) : "memory");
}
__device__ __forceinline__ void wait_async0() {
    asm volatile("s_wait_asynccnt 0x0" ::: "memory");
}

// A fragment (16x32 f16): lanes 0-15 M=0..15; VGPR i <-> K-pair i+8h.
__device__ __forceinline__ v16h load_frag_A(const unsigned* lds, int rowBase, int lane) {
    const int m = lane & 15, h = lane >> 4;
    Frag f;
#pragma unroll
    for (int i = 0; i < 8; ++i) {
        const int kp = i + 8 * h;
        f.u[i] = lds[(rowBase + m) * 17 + kp];
    }
    return f.v;
}
// B fragment (32x16 f16): lane holds N=lane&15; lanes 0-15 K=0..15, 16-31 K=16..31.
__device__ __forceinline__ v16h load_frag_B(const unsigned* lds, int rowBase, int lane) {
    const int n = lane & 15, h = lane >> 4;
    Frag f;
#pragma unroll
    for (int i = 0; i < 8; ++i)
        f.u[i] = lds[(rowBase + n) * 17 + (i + 8 * h)];
    return f.v;
}

#define WMMA(a, b, c) \
    __builtin_amdgcn_wmma_f32_16x16x32_f16(false, (a), false, (b), (short)0, (c), false, false)

// ---------------------------------------------------------------------------
// Kernel 1: c_z = W_z @ x   grid (2, 49*32, 3); fp32 sources, VGPR-staged
// ---------------------------------------------------------------------------
__global__ __launch_bounds__(256)
void conv1x1_gemm_kernel(const float* __restrict__ x,
                         const float* __restrict__ W1,
                         const float* __restrict__ W2,
                         const float* __restrict__ W3,
                         _Float16* __restrict__ c1,
                         _Float16* __restrict__ c2,
                         _Float16* __restrict__ c3) {
    __shared__ unsigned lds[LDS_TOT];
    unsigned* ldsA = lds;
    unsigned* ldsB = lds + LDSB_U32;
    const int tid = threadIdx.x, lane = tid & 31, wave = tid >> 5;
    const int wm = wave >> 1, wn = wave & 1;           // 4 x 2 waves

    const float*  Wm  = (blockIdx.z == 0) ? W1 : (blockIdx.z == 1) ? W2 : W3;
    _Float16*     out = (blockIdx.z == 0) ? c1 : (blockIdx.z == 1) ? c2 : c3;

    const int mbase = blockIdx.x * BM;
    const int b     = blockIdx.y / 49;
    const int hw0   = (blockIdx.y % 49) * BN;
    const float* xb = x + (size_t)b * C_DIM * HW + hw0;

    v8f acc[2][2] = {};
    for (int kk = 0; kk < C_DIM; kk += 32) {
        if (kk + 32 < C_DIM)
            __builtin_prefetch(xb + (size_t)(kk + 32) * HW, 0, 1);
        // ---- batch all global loads first (hide latency), then pack to LDS
        float2 av[8];                       // A: 128 x 16 pairs = 8/thread
#pragma unroll
        for (int j = 0; j < 8; ++j) {
            const int p = tid + 256 * j, m = p >> 4, kp = p & 15;
            av[j] = *(const float2*)(Wm + (size_t)(mbase + m) * C_DIM + kk + 2 * kp);
        }
        float bv0[4], bv1[4];               // B: 64 x 16 pairs = 4/thread
#pragma unroll
        for (int j = 0; j < 4; ++j) {
            const int p = tid + 256 * j, n = p & 63, kp = p >> 6;
            const float* xs = xb + (size_t)(kk + 2 * kp) * HW + n;
            bv0[j] = xs[0]; bv1[j] = xs[HW];
        }
#pragma unroll
        for (int j = 0; j < 8; ++j) {
            const int p = tid + 256 * j, m = p >> 4, kp = p & 15;
            ldsA[m * 17 + kp] = pack_f32x2_h2(av[j].x, av[j].y);
        }
#pragma unroll
        for (int j = 0; j < 4; ++j) {
            const int p = tid + 256 * j, n = p & 63, kp = p >> 6;
            ldsB[n * 17 + kp] = pack_f32x2_h2(bv0[j], bv1[j]);
        }
        __syncthreads();
        const v16h a0 = load_frag_A(ldsA, wm * 32,      lane);
        const v16h a1 = load_frag_A(ldsA, wm * 32 + 16, lane);
        const v16h b0 = load_frag_B(ldsB, wn * 32,      lane);
        const v16h b1 = load_frag_B(ldsB, wn * 32 + 16, lane);
        acc[0][0] = WMMA(a0, b0, acc[0][0]);
        acc[0][1] = WMMA(a0, b1, acc[0][1]);
        acc[1][0] = WMMA(a1, b0, acc[1][0]);
        acc[1][1] = WMMA(a1, b1, acc[1][1]);
        __syncthreads();
    }
    _Float16* outb = out + (size_t)b * C_DIM * HW + hw0;
    const int nloc = lane & 15, h = lane >> 4;
#pragma unroll
    for (int ti = 0; ti < 2; ++ti)
#pragma unroll
        for (int tj = 0; tj < 2; ++tj)
#pragma unroll
            for (int r = 0; r < 8; ++r) {
                const int mg = mbase + wm * 32 + ti * 16 + r + 8 * h;
                const int ng = wn * 32 + tj * 16 + nloc;
                outb[(size_t)mg * HW + ng] = (_Float16)acc[ti][tj][r];
            }
}

// ---------------------------------------------------------------------------
// Kernel 2: fused 2x2 max+avg pool, c1->p1, c2->p2
// ---------------------------------------------------------------------------
__global__ __launch_bounds__(256)
void pool_kernel(const _Float16* __restrict__ c1, const _Float16* __restrict__ c2,
                 _Float16* __restrict__ p1, _Float16* __restrict__ p2) {
    const _Float16* src = blockIdx.y ? c2 : c1;
    _Float16*       dst = blockIdx.y ? p2 : p1;
    const int idx = blockIdx.x * 256 + threadIdx.x;
    const int j = idx % 28;
    int t = idx / 28;
    const int i = t % 28;
    t /= 28;
    const _Float16* s = src + ((size_t)t * 56 + 2 * i) * 56 + 2 * j;
    const float v00 = (float)s[0],  v01 = (float)s[1];
    const float v10 = (float)s[56], v11 = (float)s[57];
    const float mx = fmaxf(fmaxf(v00, v01), fmaxf(v10, v11));
    dst[idx] = (_Float16)(mx + 0.25f * (v00 + v01 + v10 + v11));
}

// ---------------------------------------------------------------------------
// Kernel 3: mat[b] = p1_b (256x784) @ p2_b^T  — f16 sources, ASYNC-staged
// grid (2, 4, 32); K=784 with zero-fill tail
// ---------------------------------------------------------------------------
__global__ __launch_bounds__(256)
void gram_gemm_kernel(const _Float16* __restrict__ p1, const _Float16* __restrict__ p2,
                      float* __restrict__ mat) {
    __shared__ unsigned lds[LDS_TOT];
    unsigned* ldsA = lds;
    unsigned* ldsB = lds + LDSB_U32;
    const int tid = threadIdx.x, lane = tid & 31, wave = tid >> 5;
    const int wm = wave >> 1, wn = wave & 1;
    const int mbase = blockIdx.x * BM, nbase = blockIdx.y * BN, b = blockIdx.z;
    const unsigned long long Abase = (unsigned long long)(p1 + (size_t)b * C_DIM * PHW);
    const unsigned long long Bbase = (unsigned long long)(p2 + (size_t)b * C_DIM * PHW);

    v8f acc[2][2] = {};
    for (int kk = 0; kk < PHW; kk += 32) {
#pragma unroll
        for (int j = 0; j < 8; ++j) {          // A tile: 128 rows x 16 k-pairs
            const int p = tid + 256 * j, m = p >> 4, kp = p & 15, k = kk + 2 * kp;
            if (k < PHW)
                async_ld_b32((unsigned)(m * 17 + kp) * 4u,
                             (unsigned)(((mbase + m) * PHW + k) * 2), Abase);
            else
                ldsA[m * 17 + kp] = 0u;
        }
#pragma unroll
        for (int j = 0; j < 4; ++j) {          // B tile: 64 rows x 16 k-pairs
            const int p = tid + 256 * j, n = p & 63, kp = p >> 6, k = kk + 2 * kp;
            if (k < PHW)
                async_ld_b32((unsigned)(LDSB_U32 + n * 17 + kp) * 4u,
                             (unsigned)(((nbase + n) * PHW + k) * 2), Bbase);
            else
                ldsB[n * 17 + kp] = 0u;
        }
        wait_async0();
        __syncthreads();
        const v16h a0 = load_frag_A(ldsA, wm * 32,      lane);
        const v16h a1 = load_frag_A(ldsA, wm * 32 + 16, lane);
        const v16h b0 = load_frag_B(ldsB, wn * 32,      lane);
        const v16h b1 = load_frag_B(ldsB, wn * 32 + 16, lane);
        acc[0][0] = WMMA(a0, b0, acc[0][0]);
        acc[0][1] = WMMA(a0, b1, acc[0][1]);
        acc[1][0] = WMMA(a1, b0, acc[1][0]);
        acc[1][1] = WMMA(a1, b1, acc[1][1]);
        __syncthreads();
    }
    float* mb = mat + (size_t)b * C_DIM * C_DIM;
    const int nloc = lane & 15, h = lane >> 4;
#pragma unroll
    for (int ti = 0; ti < 2; ++ti)
#pragma unroll
        for (int tj = 0; tj < 2; ++tj)
#pragma unroll
            for (int r = 0; r < 8; ++r) {
                const int mg = mbase + wm * 32 + ti * 16 + r + 8 * h;
                const int ng = nbase + wn * 32 + tj * 16 + nloc;
                mb[(size_t)mg * C_DIM + ng] = acc[ti][tj][r];
            }
}

// ---------------------------------------------------------------------------
// Kernel 4: row softmax on [B*C, 256] -> f16 attn
// ---------------------------------------------------------------------------
__global__ __launch_bounds__(256)
void softmax_kernel(const float* __restrict__ mat, _Float16* __restrict__ attn) {
    __shared__ float red[256];
    const int row = blockIdx.x, tid = threadIdx.x;
    const float v = mat[(size_t)row * C_DIM + tid];
    red[tid] = v; __syncthreads();
    for (int s = 128; s > 0; s >>= 1) {
        if (tid < s) red[tid] = fmaxf(red[tid], red[tid + s]);
        __syncthreads();
    }
    const float mx = red[0]; __syncthreads();
    const float e = __expf(v - mx);
    red[tid] = e; __syncthreads();
    for (int s = 128; s > 0; s >>= 1) {
        if (tid < s) red[tid] += red[tid + s];
        __syncthreads();
    }
    attn[(size_t)row * C_DIM + tid] = (_Float16)(e / red[0]);
}

// ---------------------------------------------------------------------------
// Kernel 5: out = gamma * (attn_b @ c3_b) + x   grid (2, 49, 32)
// A (contiguous f16) async-staged; B (k-strided) VGPR-staged
// ---------------------------------------------------------------------------
__global__ __launch_bounds__(256)
void attn_apply_kernel(const _Float16* __restrict__ attn, const _Float16* __restrict__ c3,
                       const float* __restrict__ x, const float* __restrict__ gamma,
                       float* __restrict__ out) {
    __shared__ unsigned lds[LDS_TOT];
    unsigned* ldsA = lds;
    unsigned* ldsB = lds + LDSB_U32;
    const int tid = threadIdx.x, lane = tid & 31, wave = tid >> 5;
    const int wm = wave >> 1, wn = wave & 1;
    const int mbase = blockIdx.x * BM;
    const int b     = blockIdx.y / 49;
    const int hw0   = (blockIdx.y % 49) * BN;
    const unsigned long long Abase =
        (unsigned long long)(attn + (size_t)b * C_DIM * C_DIM);
    const unsigned short* Bb =
        (const unsigned short*)(c3 + (size_t)b * C_DIM * HW + hw0);

    v8f acc[2][2] = {};
    for (int kk = 0; kk < C_DIM; kk += 32) {
#pragma unroll
        for (int j = 0; j < 8; ++j) {          // A tile via async copy
            const int p = tid + 256 * j, m = p >> 4, kp = p & 15;
            async_ld_b32((unsigned)(m * 17 + kp) * 4u,
                         (unsigned)(((mbase + m) * C_DIM + kk + 2 * kp) * 2), Abase);
        }
        unsigned short s0[4], s1[4];           // B tile: batched strided gathers
#pragma unroll
        for (int j = 0; j < 4; ++j) {
            const int p = tid + 256 * j, n = p & 63, kp = p >> 6, k = kk + 2 * kp;
            s0[j] = Bb[(size_t)k * HW + n];
            s1[j] = Bb[(size_t)(k + 1) * HW + n];
        }
#pragma unroll
        for (int j = 0; j < 4; ++j) {
            const int p = tid + 256 * j, n = p & 63, kp = p >> 6;
            ldsB[n * 17 + kp] = pack_h2_raw(s0[j], s1[j]);
        }
        wait_async0();
        __syncthreads();
        const v16h a0 = load_frag_A(ldsA, wm * 32,      lane);
        const v16h a1 = load_frag_A(ldsA, wm * 32 + 16, lane);
        const v16h b0 = load_frag_B(ldsB, wn * 32,      lane);
        const v16h b1 = load_frag_B(ldsB, wn * 32 + 16, lane);
        acc[0][0] = WMMA(a0, b0, acc[0][0]);
        acc[0][1] = WMMA(a0, b1, acc[0][1]);
        acc[1][0] = WMMA(a1, b0, acc[1][0]);
        acc[1][1] = WMMA(a1, b1, acc[1][1]);
        __syncthreads();
    }
    const float g = gamma[0];
    const int nloc = lane & 15, h = lane >> 4;
    const size_t base = (size_t)b * C_DIM * HW + hw0;
#pragma unroll
    for (int ti = 0; ti < 2; ++ti)
#pragma unroll
        for (int tj = 0; tj < 2; ++tj)
#pragma unroll
            for (int r = 0; r < 8; ++r) {
                const int mg = mbase + wm * 32 + ti * 16 + r + 8 * h;
                const int ng = wn * 32 + tj * 16 + nloc;
                const size_t idx = base + (size_t)mg * HW + ng;
                out[idx] = g * acc[ti][tj][r] + x[idx];
            }
}

// ---------------------------------------------------------------------------
extern "C" void kernel_launch(void* const* d_in, const int* in_sizes, int n_in,
                              void* d_out, int out_size, void* d_ws, size_t ws_size,
                              hipStream_t stream) {
    const float* x     = (const float*)d_in[0];
    const float* W1    = (const float*)d_in[1];
    const float* W2    = (const float*)d_in[2];
    const float* W3    = (const float*)d_in[3];
    const float* gamma = (const float*)d_in[4];
    float* out = (float*)d_out;

    char* ws = (char*)d_ws;
    const size_t S_c = (size_t)B_DIM * C_DIM * HW  * sizeof(_Float16); // 51.4 MB
    const size_t S_p = (size_t)B_DIM * C_DIM * PHW * sizeof(_Float16); // 12.8 MB
    const size_t S_m = (size_t)B_DIM * C_DIM * C_DIM * sizeof(float);  //  8.4 MB
    _Float16* c1   = (_Float16*)(ws);
    _Float16* c2   = (_Float16*)(ws + S_c);
    _Float16* c3   = (_Float16*)(ws + 2 * S_c);
    _Float16* p1   = (_Float16*)(ws + 3 * S_c);
    _Float16* p2   = (_Float16*)(ws + 3 * S_c + S_p);
    float*    mat  = (float*)   (ws + 3 * S_c + 2 * S_p);
    _Float16* attn = (_Float16*)(ws + 3 * S_c + 2 * S_p + S_m);

    conv1x1_gemm_kernel<<<dim3(2, 49 * B_DIM, 3), 256, 0, stream>>>(
        x, W1, W2, W3, c1, c2, c3);
    pool_kernel<<<dim3((B_DIM * C_DIM * PHW) / 256, 2), 256, 0, stream>>>(
        c1, c2, p1, p2);
    gram_gemm_kernel<<<dim3(2, 4, B_DIM), 256, 0, stream>>>(p1, p2, mat);
    softmax_kernel<<<B_DIM * C_DIM, 256, 0, stream>>>(mat, attn);
    attn_apply_kernel<<<dim3(2, 49, B_DIM), 256, 0, stream>>>(
        attn, c3, x, gamma, out);
}
